// GAT_7224134992179
// MI455X (gfx1250) — compile-verified, matrix operand order checked by reading, not run on hardware
//
#include <hip/hip_runtime.h>
#include <math.h>

// ---- problem sizes (fixed by the reference) ----
#define Nn 50000
#define Ee 1600000
#define Dd 128
#define Hh 128
#define Cc 10
#define Gg 512
#define LRELU 0.05f
#define BN_EPS 1e-5f

typedef __attribute__((ext_vector_type(16))) __bf16 v16bf;
typedef __attribute__((ext_vector_type(8)))  float  v8f;

// WMMA operand viewed as 8 dwords: slot pair (2j,2j+1) == dword j
union BFrag { v16bf v; unsigned u[8]; };

__device__ __forceinline__ unsigned short f2bf_bits(float f) {
  union { float f; unsigned u; } v; v.f = f;
  return (unsigned short)((v.u + 0x7FFFu + ((v.u >> 16) & 1u)) >> 16);  // RNE
}

// ---------------- utility: zero a float region ----------------
__global__ void k_zero(float* __restrict__ p, long long n) {
  long long i = (long long)blockIdx.x * blockDim.x + threadIdx.x;
  const long long stride = (long long)gridDim.x * blockDim.x;
  for (; i < n; i += stride) p[i] = 0.0f;
}

// ---------------- one-time f32 -> bf16 conversion ----------------
__global__ void k_cvt(const float* __restrict__ s, unsigned short* __restrict__ d, long long n) {
  long long i = (long long)blockIdx.x * blockDim.x + threadIdx.x;
  const long long stride = (long long)gridDim.x * blockDim.x;
  for (; i < n; i += stride) d[i] = f2bf_bits(s[i]);
}

// ---------------- global mean / sumsq of x_in ----------------
__global__ void __launch_bounds__(256) k_stats(const float* __restrict__ x, float* __restrict__ stats) {
  __shared__ float ssum[256], ssq[256];
  const int tid = threadIdx.x;
  long long i = (long long)blockIdx.x * blockDim.x + tid;
  const long long stride = (long long)gridDim.x * blockDim.x;
  const long long total = (long long)Nn * Dd;
  float s = 0.f, q = 0.f;
  for (; i < total; i += stride) { float v = x[i]; s += v; q += v * v; }
  ssum[tid] = s; ssq[tid] = q; __syncthreads();
  for (int o = 128; o > 0; o >>= 1) {
    if (tid < o) { ssum[tid] += ssum[tid + o]; ssq[tid] += ssq[tid + o]; }
    __syncthreads();
  }
  if (tid == 0) { atomicAdd(&stats[0], ssum[0]); atomicAdd(&stats[1], ssq[0]); }
}

// fold standardization into the GEMM:  z = (x@Wt)/sigma - (mu/sigma)*rowsum(W)
__global__ void k_prep(const float* __restrict__ w, float* __restrict__ stats,
                       float* __restrict__ corr) {
  const int h = threadIdx.x;            // 128 threads
  float colsum = 0.f;
  for (int d = 0; d < Dd; ++d) colsum += w[h * Dd + d];
  const float nd = (float)((long long)Nn * Dd);
  const float s  = stats[0], ss = stats[1];
  const float mu = s / nd;
  const float var = (ss - s * s / nd) / (nd - 1.0f);   // ddof=1 (torch.std)
  const float invsig = rsqrtf(var);
  if (h == 0) stats[2] = invsig;
  corr[h] = mu * invsig * colsum;
}

// ---------------- WMMA GEMM: z[n,h] = std(x) @ fc_w^T (bf16 inputs, f32 acc) ----------------
// grid = N/16 blocks x 256 thr (8 waves); wave w owns h-tile w*16.
// Inner loop is pure global_load_b128 -> v_wmma (no per-iteration conversion VALU).
__global__ void __launch_bounds__(256) k_gat_gemm(const unsigned short* __restrict__ xbf,
                                                  const unsigned short* __restrict__ wbf,
                                                  const float* __restrict__ stats,
                                                  const float* __restrict__ corr,
                                                  float* __restrict__ z) {
  const int tid  = threadIdx.x;
  const int lane = tid & 31;
  const int hi   = lane >> 4;      // half-wave selector
  const int lm   = lane & 15;      // M for A, N for B/C/D
  const int n0   = blockIdx.x * 16;
  const int h0   = (tid >> 5) * 16;

  const unsigned* xp = (const unsigned*)xbf + (size_t)(n0 + lm) * (Dd / 2);
  const unsigned* wp = (const unsigned*)wbf + (size_t)(h0 + lm) * (Dd / 2);

  v8f c = {0.f, 0.f, 0.f, 0.f, 0.f, 0.f, 0.f, 0.f};
  const float invsig = stats[2];

  #pragma unroll
  for (int k0 = 0; k0 < Dd; k0 += 32) {
    BFrag a, b;
    #pragma unroll
    for (int j = 0; j < 8; ++j) {
      // CDNA5 16-bit A 16x32 layout: VGPR j holds K pair; lanes16-31 shift K by 8/24
      const int kk = k0 + ((j < 4) ? 0 : 16) + (hi ? 8 : 0) + 2 * (j & 3);
      a.u[j] = xp[kk >> 1];
      b.u[j] = wp[kk >> 1];
    }
    if (k0 + 32 < Dd)
      __builtin_prefetch(xp + ((k0 + 32) >> 1), 0, 1);   // global_prefetch_b8
    c = __builtin_amdgcn_wmma_f32_16x16x32_bf16(false, a.v, false, b.v, (short)0, c, false, false);
  }
  #pragma unroll
  for (int r = 0; r < 8; ++r) {
    const int m = r + hi * 8;                 // C/D layout: lanes16-31 hold M+8
    const int h = h0 + lm;
    z[(size_t)(n0 + m) * Hh + h] = c[r] * invsig - corr[h];
  }
}

// ---------------- s1 = z@a1, s2 = z@a2 (one wave per node) ----------------
__global__ void __launch_bounds__(256) k_s12(const float* __restrict__ z,
                                             const float* __restrict__ a1,
                                             const float* __restrict__ a2,
                                             float* __restrict__ s1, float* __restrict__ s2) {
  const int lane = threadIdx.x & 31;
  const int n = blockIdx.x * 8 + (threadIdx.x >> 5);
  if (n >= Nn) return;
  float p1 = 0.f, p2 = 0.f;
  for (int i = lane; i < Hh; i += 32) {
    const float zv = z[(size_t)n * Hh + i];
    p1 += zv * a1[i]; p2 += zv * a2[i];
  }
  for (int o = 16; o > 0; o >>= 1) { p1 += __shfl_xor(p1, o); p2 += __shfl_xor(p2, o); }
  if (lane == 0) { s1[n] = p1; s2[n] = p2; }
}

// ---------------- edge pass 1: h = exp(lrelu(s1[src]+s2[dst])), hsum[src] += h ----------------
__global__ void __launch_bounds__(256) k_edge1(const int* __restrict__ src, const int* __restrict__ dst,
                                               const float* __restrict__ s1, const float* __restrict__ s2,
                                               float* __restrict__ hbuf, float* __restrict__ hsum) {
  long long i = (long long)blockIdx.x * blockDim.x + threadIdx.x;
  const long long stride = (long long)gridDim.x * blockDim.x;
  for (; i < Ee; i += stride) {
    const int s = src[i], d = dst[i];
    float e = s1[s] + s2[d];
    e = (e >= 0.f) ? e : LRELU * e;
    const float h = __expf(e);
    hbuf[i] = h;
    atomicAdd(&hsum[s], h);
  }
}

// ---------------- edge pass 2: out[src] += alpha * z[dst] (one wave per edge) ----------------
__global__ void __launch_bounds__(256) k_edge2(const int* __restrict__ src, const int* __restrict__ dst,
                                               const float* __restrict__ hbuf, const float* __restrict__ hsum,
                                               const float* __restrict__ z, float* __restrict__ gout) {
  const int lane = threadIdx.x & 31;
  const long long e = (long long)blockIdx.x * 8 + (threadIdx.x >> 5);
  if (e >= Ee) return;
  const int s = src[e], d = dst[e];
  const float alpha = hbuf[e] / hsum[s];
  const float4 zv = *(const float4*)(z + (size_t)d * Hh + lane * 4);
  float* o = gout + (size_t)s * Hh + lane * 4;   // z & gout are L2-resident (25.6MB each)
  atomicAdd(o + 0, alpha * zv.x);
  atomicAdd(o + 1, alpha * zv.y);
  atomicAdd(o + 2, alpha * zv.z);
  atomicAdd(o + 3, alpha * zv.w);
}

// ---------------- relu + sum pooling per graph ----------------
__global__ void __launch_bounds__(256) k_pool(const float* __restrict__ gout, const int* __restrict__ idx,
                                              float* __restrict__ pool) {
  const int lane = threadIdx.x & 31;
  const int n = blockIdx.x * 8 + (threadIdx.x >> 5);
  if (n >= Nn) return;
  const int g = idx[n];
  const float4 v = *(const float4*)(gout + (size_t)n * Hh + lane * 4);
  float* p = pool + (size_t)g * Hh + lane * 4;
  atomicAdd(p + 0, fmaxf(v.x, 0.f));
  atomicAdd(p + 1, fmaxf(v.y, 0.f));
  atomicAdd(p + 2, fmaxf(v.z, 0.f));
  atomicAdd(p + 3, fmaxf(v.w, 0.f));
}

// ---------------- BatchNorm1d (batch stats, biased var) ----------------
__global__ void k_bn(const float* __restrict__ pooled, const float* __restrict__ gamma,
                     const float* __restrict__ beta, float* __restrict__ xb) {
  const int h = threadIdx.x;     // 128 threads
  float s = 0.f, q = 0.f;
  for (int g = 0; g < Gg; ++g) { const float v = pooled[g * Hh + h]; s += v; q += v * v; }
  const float mu = s / (float)Gg;
  const float var = q / (float)Gg - mu * mu;
  const float inv = rsqrtf(var + BN_EPS);
  const float ga = gamma[h], be = beta[h];
  for (int g = 0; g < Gg; ++g)
    xb[g * Hh + h] = (pooled[g * Hh + h] - mu) * inv * ga + be;
}

// ---------------- fc1 via WMMA: relu(xb @ fc1_w^T + b) ----------------
__global__ void __launch_bounds__(256) k_fc1(const unsigned short* __restrict__ xbf,
                                             const unsigned short* __restrict__ wbf,
                                             const float* __restrict__ bias, float* __restrict__ out) {
  const int tid  = threadIdx.x;
  const int lane = tid & 31;
  const int hi   = lane >> 4;
  const int lm   = lane & 15;
  const int n0   = blockIdx.x * 16;        // 32 blocks cover G=512 rows
  const int h0   = (tid >> 5) * 16;

  const unsigned* xp = (const unsigned*)xbf + (size_t)(n0 + lm) * (Hh / 2);
  const unsigned* wp = (const unsigned*)wbf + (size_t)(h0 + lm) * (Hh / 2);

  v8f c = {0.f, 0.f, 0.f, 0.f, 0.f, 0.f, 0.f, 0.f};
  #pragma unroll
  for (int k0 = 0; k0 < Hh; k0 += 32) {
    BFrag a, b;
    #pragma unroll
    for (int j = 0; j < 8; ++j) {
      const int kk = k0 + ((j < 4) ? 0 : 16) + (hi ? 8 : 0) + 2 * (j & 3);
      a.u[j] = xp[kk >> 1];
      b.u[j] = wp[kk >> 1];
    }
    c = __builtin_amdgcn_wmma_f32_16x16x32_bf16(false, a.v, false, b.v, (short)0, c, false, false);
  }
  #pragma unroll
  for (int r = 0; r < 8; ++r) {
    const int m = r + hi * 8;
    const int h = h0 + lm;
    out[(size_t)(n0 + m) * Hh + h] = fmaxf(c[r] + bias[h], 0.f);
  }
}

// ---------------- fc2 + log_softmax (one wave per row, C=10) ----------------
__global__ void __launch_bounds__(256) k_fc2(const float* __restrict__ x, const float* __restrict__ w,
                                             const float* __restrict__ b, float* __restrict__ out) {
  const int lane = threadIdx.x & 31;
  const int g = blockIdx.x * 8 + (threadIdx.x >> 5);
  if (g >= Gg) return;
  float dot = -INFINITY;
  if (lane < Cc) {
    float acc = b[lane];
    for (int d = 0; d < Hh; ++d) acc += x[(size_t)g * Hh + d] * w[lane * Hh + d];
    dot = acc;
  }
  float m = dot;
  for (int o = 16; o > 0; o >>= 1) m = fmaxf(m, __shfl_xor(m, o));
  float ex = (lane < Cc) ? __expf(dot - m) : 0.f;
  float se = ex;
  for (int o = 16; o > 0; o >>= 1) se += __shfl_xor(se, o);
  if (lane < Cc) out[(size_t)g * Cc + lane] = dot - m - logf(se);
}

extern "C" void kernel_launch(void* const* d_in, const int* in_sizes, int n_in,
                              void* d_out, int out_size, void* d_ws, size_t ws_size,
                              hipStream_t stream) {
  const float* x_in  = (const float*)d_in[0];
  const int*   eidx  = (const int*)  d_in[1];   // [2,E] row-major: row0=src, row1=dst
  const int*   idx   = (const int*)  d_in[2];
  const float* fc_w  = (const float*)d_in[3];
  const float* a1    = (const float*)d_in[4];
  const float* a2    = (const float*)d_in[5];
  const float* fc1_w = (const float*)d_in[6];
  const float* fc1_b = (const float*)d_in[7];
  const float* fc2_w = (const float*)d_in[8];
  const float* fc2_b = (const float*)d_in[9];
  const float* gamma = (const float*)d_in[10];
  const float* beta  = (const float*)d_in[11];
  const int* src = eidx;
  const int* dst = eidx + Ee;

  float* ws   = (float*)d_ws;
  float* STAT = ws;                              // 16 floats
  float* CORR = ws + 64;                         // 128 floats
  float* Z    = ws + 256;                        // N*H f32
  float* S1   = Z + (size_t)Nn * Hh;             // N
  float* S2   = S1 + Nn;                         // N
  float* HB   = S2 + Nn;                         // E
  float* HS   = HB + Ee;                         // N
  float* GOUT = HS + Nn;                         // N*H f32
  float* POOL = GOUT + (size_t)Nn * Hh;          // G*H
  float* XB   = POOL + (size_t)Gg * Hh;          // G*H
  float* FC1O = XB + (size_t)Gg * Hh;            // G*H
  // bf16 staging buffers (ushort counts; float-slot sizes are count/2)
  unsigned short* XBF  = (unsigned short*)(FC1O + (size_t)Gg * Hh);   // N*D bf16 (12.8MB)
  unsigned short* WBF  = XBF  + (size_t)Nn * Dd;                      // H*D bf16
  unsigned short* W1BF = WBF  + (size_t)Hh * Dd;                      // H*H bf16
  unsigned short* XBBF = W1BF + (size_t)Hh * Hh;                      // G*H bf16

  (void)in_sizes; (void)n_in; (void)out_size; (void)ws_size;

  // zero the accumulators (harness poisons ws with 0xAA)
  k_zero<<<1,    256, 0, stream>>>(STAT, 16);
  k_zero<<<256,  256, 0, stream>>>(HS, Nn);
  k_zero<<<2048, 256, 0, stream>>>(GOUT, (long long)Nn * Hh);
  k_zero<<<64,   256, 0, stream>>>(POOL, (long long)Gg * Hh);

  // one-time bf16 staging (each element converted exactly once)
  k_cvt<<<2048, 256, 0, stream>>>(x_in,  XBF,  (long long)Nn * Dd);
  k_cvt<<<64,   256, 0, stream>>>(fc_w,  WBF,  (long long)Hh * Dd);
  k_cvt<<<64,   256, 0, stream>>>(fc1_w, W1BF, (long long)Hh * Hh);

  k_stats<<<1024, 256, 0, stream>>>(x_in, STAT);
  k_prep<<<1, 128, 0, stream>>>(fc_w, STAT, CORR);
  k_gat_gemm<<<Nn / 16, 256, 0, stream>>>(XBF, WBF, STAT, CORR, Z);
  k_s12<<<Nn / 8, 256, 0, stream>>>(Z, a1, a2, S1, S2);
  k_edge1<<<4096, 256, 0, stream>>>(src, dst, S1, S2, HB, HS);
  k_edge2<<<Ee / 8, 256, 0, stream>>>(src, dst, HB, HS, Z, GOUT);
  k_pool<<<Nn / 8, 256, 0, stream>>>(GOUT, idx, POOL);
  k_bn<<<1, 128, 0, stream>>>(POOL, gamma, beta, XB);
  k_cvt<<<64, 256, 0, stream>>>(XB, XBBF, (long long)Gg * Hh);
  k_fc1<<<Gg / 16, 256, 0, stream>>>(XBBF, W1BF, fc1_b, FC1O);
  k_fc2<<<Gg / 8, 256, 0, stream>>>(FC1O, fc2_w, fc2_b, (float*)d_out);
}
